// MultiheadDiffAttn_74131135529005
// MI455X (gfx1250) — matrix operand order repaired and checked by reading, compiled, not run on hardware
//
#include <hip/hip_runtime.h>
#include <hip/hip_bf16.h>

typedef __bf16 bf16_t;
typedef bf16_t v16bf __attribute__((ext_vector_type(16)));
typedef float  v8f   __attribute__((ext_vector_type(8)));

// ---------------------------------------------------------------------------
// WMMA fragment helpers (wave32, V_WMMA_F32_16X16X32_BF16)
//
// A (16x32, row-major src): lane l -> row = l&15, K = (l>>4)*16 + {0..15}
//                           (contiguous halves, same k order as B)
// B (32x16): lane l -> col = l&15, K = (l>>4)*16 + {0..15} (contiguous halves,
//            same k order as the A fragment)
// C/D (16x16 f32): vgpr i, lane l -> row = i + 8*(l>>4), col = l&15
//
// All fragments below are contiguous 16B-aligned loads (2 x b128 per lane).
// ---------------------------------------------------------------------------

__device__ __forceinline__ v8f v8f_zero() {
  v8f z = {0.f, 0.f, 0.f, 0.f, 0.f, 0.f, 0.f, 0.f};
  return z;
}

__device__ __forceinline__ v8f wmma_bf16(v16bf a, v16bf b, v8f c) {
  return __builtin_amdgcn_wmma_f32_16x16x32_bf16(
      /*neg_a=*/false, a, /*neg_b=*/false, b,
      /*c_mod=*/(short)0, c, /*reuse_a=*/false, /*reuse_b=*/false);
}

// A fragment from row-major matrix (rows = M, row stride ld elements).
// Works for both global and LDS pointers (addrspace inferred after inlining).
__device__ __forceinline__ v16bf wmma_load_a(const bf16_t* p, int ld) {
  const int lane = (int)(threadIdx.x & 31u);
  const int row  = lane & 15;
  const int kb   = (lane >> 4) * 16;
  const bf16_t* r = p + (size_t)row * (size_t)ld;
  union { uint4 u[2]; v16bf v; } t;
  t.u[0] = *(const uint4*)(r + kb);        // K = kb .. kb+7
  t.u[1] = *(const uint4*)(r + kb + 8);    // K = kb+8 .. kb+15
  return t.v;
}

// B fragment from an N-major matrix (element (k,n) at p[n*ld + k]):
// contiguous 32B per lane. Used for k^T tiles, transposed weights, v^T tiles.
__device__ __forceinline__ v16bf wmma_load_b_nk(const bf16_t* p, int ld) {
  const int lane = (int)(threadIdx.x & 31u);
  const int col  = lane & 15;
  const int kb   = (lane >> 4) * 16;
  const bf16_t* r = p + (size_t)col * (size_t)ld + kb;
  union { uint4 u[2]; v16bf v; } t;
  t.u[0] = *(const uint4*)(r);
  t.u[1] = *(const uint4*)(r + 8);
  return t.v;
}

// ---------------------------------------------------------------------------
// Kernel 1a: fp32 -> bf16 conversion, 4 elements/thread (grid-stride)
// ---------------------------------------------------------------------------
__global__ void convert_bf16x4_kernel(const float* __restrict__ src,
                                      bf16_t* __restrict__ dst, int n4) {
  int i = (int)(blockIdx.x * blockDim.x + threadIdx.x);
  const int stride = (int)(gridDim.x * blockDim.x);
  for (; i < n4; i += stride) {
    const float4 f = ((const float4*)src)[i];
    union { bf16_t b[4]; uint2 u; } t;
    t.b[0] = (bf16_t)f.x; t.b[1] = (bf16_t)f.y;
    t.b[2] = (bf16_t)f.z; t.b[3] = (bf16_t)f.w;
    ((uint2*)dst)[i] = t.u;
  }
}

// ---------------------------------------------------------------------------
// Kernel 1b: weight transpose + convert: W [K=1024, N=1024] f32 -> W^T [N, K]
// bf16. LDS-tiled 32x32 so both global streams stay coalesced.
// grid = (32, 32), block = 256 (32x8)
// ---------------------------------------------------------------------------
__global__ __launch_bounds__(256)
void transpose_convert_kernel(const float* __restrict__ src,
                              bf16_t* __restrict__ dst) {
  __shared__ float tile[32][33];
  const int tx = (int)(threadIdx.x & 31u);
  const int ty = (int)(threadIdx.x >> 5);     // 0..7
  const int n0 = (int)blockIdx.x * 32;
  const int k0 = (int)blockIdx.y * 32;
#pragma unroll
  for (int r = 0; r < 4; ++r)
    tile[ty * 4 + r][tx] = src[(size_t)(k0 + ty * 4 + r) * 1024 + n0 + tx];
  __syncthreads();
#pragma unroll
  for (int r = 0; r < 4; ++r)
    dst[(size_t)(n0 + ty * 4 + r) * 1024 + k0 + tx] =
        (bf16_t)tile[tx][ty * 4 + r];
}

// ---------------------------------------------------------------------------
// Kernel 2: lambda_full = exp(sum lq1*lk1) - exp(sum lq2*lk2) + LAMBDA_INIT
// HEAD_DIM == 32 -> one wave, one element per lane.
// ---------------------------------------------------------------------------
__global__ void lambda_kernel(const float* __restrict__ lq1, const float* __restrict__ lk1,
                              const float* __restrict__ lq2, const float* __restrict__ lk2,
                              float* __restrict__ out) {
  int lane = (int)threadIdx.x;  // blockDim.x == 32
  float s1 = lq1[lane] * lk1[lane];
  float s2 = lq2[lane] * lk2[lane];
#pragma unroll
  for (int m = 1; m < 32; m <<= 1) {
    s1 += __shfl_xor(s1, m, 32);
    s2 += __shfl_xor(s2, m, 32);
  }
  if (lane == 0) {
    const float lambda_init = 0.8f - 0.6f * __expf(-1.8f);
    out[0] = __expf(s1) - __expf(s2) + lambda_init;
  }
}

// ---------------------------------------------------------------------------
// Kernel 3: fused QKV projection. grid = (N/64, M/128, 3), block = 256 (8 waves)
// Each wave computes a 16x64 tile with 4 WMMA accumulators, K loop of 32.
// Weights come in transposed ([N,K] bf16) so B fragments are contiguous.
// q is scaled by D^-0.5; outputs are stored head-permuted:
//   q,k: [B, 2H, T, 32]   v^T: [B, H, 64, T]
// ---------------------------------------------------------------------------
__global__ __launch_bounds__(256)
void qkv_gemm_kernel(const bf16_t* __restrict__ x,
                     const bf16_t* __restrict__ WqT, const bf16_t* __restrict__ WkT,
                     const bf16_t* __restrict__ WvT,
                     bf16_t* __restrict__ qws, bf16_t* __restrict__ kws,
                     bf16_t* __restrict__ vwsT) {
  const int proj = (int)blockIdx.z;
  const bf16_t* WT = (proj == 0) ? WqT : ((proj == 1) ? WkT : WvT);
  const int n0   = (int)blockIdx.x * 64;
  const int wave = (int)(threadIdx.x >> 5);
  const int m0   = (int)blockIdx.y * 128 + wave * 16;

  const bf16_t* arow = x + (size_t)m0 * 1024;
  const bf16_t* brow = WT + (size_t)n0 * 1024;

  v8f acc[4] = {v8f_zero(), v8f_zero(), v8f_zero(), v8f_zero()};
  for (int k0 = 0; k0 < 1024; k0 += 32) {
    __builtin_prefetch(arow + k0 + 64, 0, 3);
    v16bf a = wmma_load_a(arow + k0, 1024);
#pragma unroll
    for (int j = 0; j < 4; ++j) {
      v16bf b = wmma_load_b_nk(brow + (size_t)(j * 16) * 1024 + k0, 1024);
      acc[j] = wmma_bf16(a, b, acc[j]);
    }
  }

  const int lane = (int)(threadIdx.x & 31u);
  const int colb = lane & 15;
  const int half = lane >> 4;
  const float scaling = 0.17677669529663687f;  // 32^-0.5
#pragma unroll
  for (int j = 0; j < 4; ++j) {
#pragma unroll
    for (int i = 0; i < 8; ++i) {
      const int r = m0 + i + 8 * half;        // token row in [0, 4096)
      const int c = n0 + j * 16 + colb;       // embed col in [0, 1024)
      const int bb = r >> 10, t = r & 1023;
      const float val = acc[j][i];
      if (proj == 0) {
        const int h2 = c >> 5, d = c & 31;
        qws[(((size_t)(bb * 32 + h2)) * 1024 + t) * 32 + d] = (bf16_t)(val * scaling);
      } else if (proj == 1) {
        const int h2 = c >> 5, d = c & 31;
        kws[(((size_t)(bb * 32 + h2)) * 1024 + t) * 32 + d] = (bf16_t)val;
      } else {
        const int h = c >> 6, dd = c & 63;   // v^T: [B,H,64,T]
        vwsT[(((size_t)(bb * 16 + h)) * 64 + dd) * 1024 + t] = (bf16_t)val;
      }
    }
  }
}

// ---------------------------------------------------------------------------
// Kernel 4: fused differential attention.
// grid = (B*H = 64, T/128 = 8), block = 256 (8 independent waves).
// Each wave owns 16 query rows of head pair (2h, 2h+1):
//   pass 1: online softmax stats (m, l) for both components via WMMA scores
//   pass 2: recompute score tiles, p0 - lambda*p1 -> d_out attn_weights,
//           reshape through LDS, accumulate (p0-lam*p1) @ v via WMMA,
//   epilogue: RMS norm over 2D=64, * subln_w * (1-LAMBDA_INIT) -> bf16 buffer
// ---------------------------------------------------------------------------
__global__ __launch_bounds__(256)
void diff_attn_kernel(const bf16_t* __restrict__ qws, const bf16_t* __restrict__ kws,
                      const bf16_t* __restrict__ vwsT, const float* __restrict__ lam_ws,
                      const float* __restrict__ subw,
                      float* __restrict__ attnw_out, bf16_t* __restrict__ attnbuf) {
  __shared__ bf16_t lds_w[8][16 * 32];   // per-wave 16x32 bf16 staging tile (8 KB)

  const int bh   = (int)blockIdx.x;        // 0..63
  const int b    = bh >> 4;
  const int h    = bh & 15;
  const int wave = (int)(threadIdx.x >> 5);
  const int lane = (int)(threadIdx.x & 31u);
  const int q0   = (int)blockIdx.y * 128 + wave * 16;
  const float lam = lam_ws[0];

  const bf16_t* qb0 = qws + (((size_t)(b * 32 + 2 * h    )) * 1024 + q0) * 32;
  const bf16_t* qb1 = qws + (((size_t)(b * 32 + 2 * h + 1)) * 1024 + q0) * 32;
  const bf16_t* kb0 = kws + (((size_t)(b * 32 + 2 * h    )) * 1024) * 32;
  const bf16_t* kb1 = kws + (((size_t)(b * 32 + 2 * h + 1)) * 1024) * 32;
  const bf16_t* vbT = vwsT + (((size_t)(b * 16 + h)) * 64) * 1024;  // [64, 1024]

  const v16bf a0 = wmma_load_a(qb0, 32);
  const v16bf a1 = wmma_load_a(qb1, 32);

  float m0[8], l0[8], m1[8], l1[8];
#pragma unroll
  for (int i = 0; i < 8; ++i) { m0[i] = -3.0e38f; l0[i] = 0.f; m1[i] = -3.0e38f; l1[i] = 0.f; }

  // ---- pass 1: online softmax statistics ----
  for (int kk = 0; kk < 1024; kk += 16) {
    __builtin_prefetch(kb0 + (size_t)(kk + 16) * 32, 0, 3);
    __builtin_prefetch(kb1 + (size_t)(kk + 16) * 32, 0, 3);
    v8f s0 = wmma_bf16(a0, wmma_load_b_nk(kb0 + (size_t)kk * 32, 32), v8f_zero());
#pragma unroll
    for (int i = 0; i < 8; ++i) {
      float t = s0[i];
      t = fmaxf(t, __shfl_xor(t, 1, 32));
      t = fmaxf(t, __shfl_xor(t, 2, 32));
      t = fmaxf(t, __shfl_xor(t, 4, 32));
      t = fmaxf(t, __shfl_xor(t, 8, 32));
      const float mn = fmaxf(m0[i], t);
      float e = __expf(s0[i] - mn);
      e += __shfl_xor(e, 1, 32);
      e += __shfl_xor(e, 2, 32);
      e += __shfl_xor(e, 4, 32);
      e += __shfl_xor(e, 8, 32);
      l0[i] = l0[i] * __expf(m0[i] - mn) + e;
      m0[i] = mn;
    }
    v8f s1 = wmma_bf16(a1, wmma_load_b_nk(kb1 + (size_t)kk * 32, 32), v8f_zero());
#pragma unroll
    for (int i = 0; i < 8; ++i) {
      float t = s1[i];
      t = fmaxf(t, __shfl_xor(t, 1, 32));
      t = fmaxf(t, __shfl_xor(t, 2, 32));
      t = fmaxf(t, __shfl_xor(t, 4, 32));
      t = fmaxf(t, __shfl_xor(t, 8, 32));
      const float mn = fmaxf(m1[i], t);
      float e = __expf(s1[i] - mn);
      e += __shfl_xor(e, 1, 32);
      e += __shfl_xor(e, 2, 32);
      e += __shfl_xor(e, 4, 32);
      e += __shfl_xor(e, 8, 32);
      l1[i] = l1[i] * __expf(m1[i] - mn) + e;
      m1[i] = mn;
    }
  }
  float li0[8], li1[8];
#pragma unroll
  for (int i = 0; i < 8; ++i) { li0[i] = 1.0f / l0[i]; li1[i] = 1.0f / l1[i]; }

  // ---- pass 2: probabilities, differential combine, W@v ----
  v8f attn_acc[4] = {v8f_zero(), v8f_zero(), v8f_zero(), v8f_zero()};
  float* awout = attnw_out + (((size_t)bh) * 1024 + q0) * 1024;
  const int colb = lane & 15;
  const int half = lane >> 4;
  bf16_t* myw = &lds_w[wave][0];

  for (int kk = 0; kk < 1024; kk += 32) {
#pragma unroll
    for (int sub = 0; sub < 2; ++sub) {
      const int kcol = kk + sub * 16;
      v8f s0 = wmma_bf16(a0, wmma_load_b_nk(kb0 + (size_t)kcol * 32, 32), v8f_zero());
      v8f s1 = wmma_bf16(a1, wmma_load_b_nk(kb1 + (size_t)kcol * 32, 32), v8f_zero());
#pragma unroll
      for (int i = 0; i < 8; ++i) {
        const float p0 = __expf(s0[i] - m0[i]) * li0[i];
        const float p1 = __expf(s1[i] - m1[i]) * li1[i];
        const float w  = p0 - lam * p1;
        const int r = i + 8 * half;
        awout[(size_t)r * 1024 + kcol + colb] = w;       // attn_weights output
        myw[r * 32 + sub * 16 + colb] = (bf16_t)w;       // stage for A-layout reload
      }
    }
    // LDS reshape: C-layout (above) -> A-layout fragment (in-order DS pipeline,
    // wave-private slice, no cross-wave sharing)
    v16bf wa = wmma_load_a(myw, 32);
#pragma unroll
    for (int j = 0; j < 4; ++j) {
      v16bf bvf = wmma_load_b_nk(vbT + (size_t)(j * 16) * 1024 + kk, 1024);
      attn_acc[j] = wmma_bf16(wa, bvf, attn_acc[j]);
    }
  }

  // ---- epilogue: RMS norm over 2D=64, * subln_w * (1 - LAMBDA_INIT) ----
  const float lambda_init = 0.8f - 0.6f * __expf(-1.8f);
  const float post = 1.0f - lambda_init;
  float sw[4];
#pragma unroll
  for (int j = 0; j < 4; ++j) sw[j] = subw[j * 16 + colb];

#pragma unroll
  for (int i = 0; i < 8; ++i) {
    float ss = 0.f;
#pragma unroll
    for (int j = 0; j < 4; ++j) ss += attn_acc[j][i] * attn_acc[j][i];
    ss += __shfl_xor(ss, 1, 32);
    ss += __shfl_xor(ss, 2, 32);
    ss += __shfl_xor(ss, 4, 32);
    ss += __shfl_xor(ss, 8, 32);
    const float sc = rsqrtf(ss * (1.0f / 64.0f) + 1e-5f) * post;
    const int t = q0 + i + 8 * half;
    bf16_t* row = attnbuf + ((size_t)b * 1024 + t) * 1024 + h * 64;
#pragma unroll
    for (int j = 0; j < 4; ++j)
      row[j * 16 + colb] = (bf16_t)(attn_acc[j][i] * sc * sw[j]);
  }
}

// ---------------------------------------------------------------------------
// Kernel 5: output projection  out = attn @ Wo  (fp32 out). Same tiling as QKV,
// Wo comes in transposed so B fragments are contiguous b128 loads.
// ---------------------------------------------------------------------------
__global__ __launch_bounds__(256)
void out_gemm_kernel(const bf16_t* __restrict__ ab, const bf16_t* __restrict__ WoT,
                     float* __restrict__ out) {
  const int n0   = (int)blockIdx.x * 64;
  const int wave = (int)(threadIdx.x >> 5);
  const int m0   = (int)blockIdx.y * 128 + wave * 16;

  const bf16_t* arow = ab + (size_t)m0 * 1024;
  const bf16_t* brow = WoT + (size_t)n0 * 1024;

  v8f acc[4] = {v8f_zero(), v8f_zero(), v8f_zero(), v8f_zero()};
  for (int k0 = 0; k0 < 1024; k0 += 32) {
    __builtin_prefetch(arow + k0 + 64, 0, 3);
    v16bf a = wmma_load_a(arow + k0, 1024);
#pragma unroll
    for (int j = 0; j < 4; ++j) {
      v16bf b = wmma_load_b_nk(brow + (size_t)(j * 16) * 1024 + k0, 1024);
      acc[j] = wmma_bf16(a, b, acc[j]);
    }
  }
  const int lane = (int)(threadIdx.x & 31u);
  const int colb = lane & 15;
  const int half = lane >> 4;
#pragma unroll
  for (int j = 0; j < 4; ++j)
#pragma unroll
    for (int i = 0; i < 8; ++i)
      out[((size_t)(m0 + i + 8 * half)) * 1024 + n0 + j * 16 + colb] = acc[j][i];
}

// ---------------------------------------------------------------------------
// Host launcher
// ---------------------------------------------------------------------------
extern "C" void kernel_launch(void* const* d_in, const int* in_sizes, int n_in,
                              void* d_out, int out_size, void* d_ws, size_t ws_size,
                              hipStream_t stream) {
  const float* x    = (const float*)d_in[0];
  const float* Wq   = (const float*)d_in[1];
  const float* Wk   = (const float*)d_in[2];
  const float* Wv   = (const float*)d_in[3];
  const float* Wo   = (const float*)d_in[4];
  const float* lq1  = (const float*)d_in[5];
  const float* lk1  = (const float*)d_in[6];
  const float* lq2  = (const float*)d_in[7];
  const float* lk2  = (const float*)d_in[8];
  const float* subw = (const float*)d_in[9];

  float* out   = (float*)d_out;                       // [4,1024,1024] fp32
  float* attnw = out + (size_t)4 * 1024 * 1024;       // [4,16,1024,1024] fp32

  // workspace layout (48 MB + 256 B)
  char* ws = (char*)d_ws;
  float*  lam = (float*)ws;
  bf16_t* xb  = (bf16_t*)(ws + 256);                  // [4096,1024]
  bf16_t* wqb = xb  + (size_t)4096 * 1024;            // [1024,1024] (transposed) each
  bf16_t* wkb = wqb + (size_t)1024 * 1024;
  bf16_t* wvb = wkb + (size_t)1024 * 1024;
  bf16_t* wob = wvb + (size_t)1024 * 1024;
  bf16_t* qb  = wob + (size_t)1024 * 1024;            // [4,32,1024,32]
  bf16_t* kbf = qb  + (size_t)4096 * 1024;            // [4,32,1024,32]
  bf16_t* vbf = kbf + (size_t)4096 * 1024;            // [4,16,64,1024] (v^T)
  bf16_t* abf = vbf + (size_t)4096 * 1024;            // [4096,1024]

  convert_bf16x4_kernel<<<1024, 256, 0, stream>>>(x, xb, 1024 * 1024);
  transpose_convert_kernel<<<dim3(32, 32), 256, 0, stream>>>(Wq, wqb);
  transpose_convert_kernel<<<dim3(32, 32), 256, 0, stream>>>(Wk, wkb);
  transpose_convert_kernel<<<dim3(32, 32), 256, 0, stream>>>(Wv, wvb);
  transpose_convert_kernel<<<dim3(32, 32), 256, 0, stream>>>(Wo, wob);
  lambda_kernel<<<1, 32, 0, stream>>>(lq1, lk1, lq2, lk2, lam);

  qkv_gemm_kernel<<<dim3(16, 32, 3), 256, 0, stream>>>(xb, wqb, wkb, wvb, qb, kbf, vbf);
  diff_attn_kernel<<<dim3(64, 8), 256, 0, stream>>>(qb, kbf, vbf, lam, subw, attnw, abf);
  out_gemm_kernel<<<dim3(16, 32), 256, 0, stream>>>(abf, wob, out);
}
